// CausalWanSelfAttention_25580825215074
// MI455X (gfx1250) — matrix operand order repaired and checked by reading, not compiled, not run on hardware
//
#include <hip/hip_runtime.h>

// ---------------------------------------------------------------------------
// CausalWanSelfAttention for MI455X (gfx1250): f16 WMMA pipeline, f32 accum,
// async global->LDS double-buffered attention tiles.
// ---------------------------------------------------------------------------

typedef __attribute__((ext_vector_type(16))) _Float16 v16h;
typedef __attribute__((ext_vector_type(8)))  _Float16 v8h;
typedef __attribute__((ext_vector_type(8)))  float    v8f;

#ifndef __has_builtin
#define __has_builtin(x) 0
#endif

#if __has_builtin(__builtin_amdgcn_global_load_async_to_lds_b128)
#define ASYNC_LDS 1
#else
#define ASYNC_LDS 0
#endif

// Pointer types matching the async builtin's prototype:
//   param0: int __attribute__((vector_size(16))) __device__ *   (global src)
//   param1: LDS destination (address_space(3))
typedef int v4i __attribute__((vector_size(16)));
typedef __attribute__((address_space(1))) v4i gas_v4i;
typedef __attribute__((address_space(3))) v4i las_v4i;

// Fixed problem shape (from setup_inputs: f=3,h=30,w=52,dim=1536,heads=12)
constexpr int DIM  = 1536;
constexpr int NH   = 12;
constexpr int DH   = 128;          // head dim
constexpr int CC   = 64;           // DH/2 rope pairs per head
constexpr int C0   = 22;           // c - 2*(c/3)
constexpr int C1   = 21;           // c/3
constexpr int WN   = 52;           // w
constexpr int HW   = 1560;         // h*w tokens per frame
constexpr int LTOK = 4680;         // f*h*w
constexpr int LP   = 4736;         // padded to multiple of 64
constexpr int QW   = 4;            // q-tile waves per attention block

// Workspace layout (bytes)
constexpr size_t SZ_XH   = (size_t)LP * DIM * 2;        // x in f16, padded
constexpr size_t SZ_W    = (size_t)DIM * DIM * 2;       // one weight in f16
constexpr size_t SZ_RAW  = (size_t)LP * DIM * 4;        // f32 projection
constexpr size_t SZ_H    = (size_t)LP * DIM * 2;        // f16 L x DIM
constexpr size_t OFF_XH    = 0;
constexpr size_t OFF_WQH   = OFF_XH   + SZ_XH;
constexpr size_t OFF_WKH   = OFF_WQH  + SZ_W;
constexpr size_t OFF_WVH   = OFF_WKH  + SZ_W;
constexpr size_t OFF_WOH   = OFF_WVH  + SZ_W;
constexpr size_t OFF_QRAW  = OFF_WOH  + SZ_W;
constexpr size_t OFF_KRAW  = OFF_QRAW + SZ_RAW;
constexpr size_t OFF_QH    = OFF_KRAW + SZ_RAW;
constexpr size_t OFF_KH    = OFF_QH   + SZ_H;
constexpr size_t OFF_VT    = OFF_KH   + SZ_H;           // [DIM][LP] f16 (vT)
constexpr size_t OFF_ATTNH = OFF_VT   + SZ_H;           // attn out f16 [LP][DIM]

// ---------------------------------------------------------------------------
// WMMA helpers
// ---------------------------------------------------------------------------
__device__ __forceinline__ v8f wmma_f16(v16h a, v16h b, v8f c) {
  return __builtin_amdgcn_wmma_f32_16x16x32_f16(false, a, false, b,
                                                (short)0, c, false, false);
}

// A-fragment (16x32 f16): lane m = l&15, off=(l>>4)*8
// element i in [0,16) -> K = 2*off + i (16 contiguous K per lane half)
__device__ __forceinline__ v16h load_a16(const _Float16* p, int off) {
  return *(const v16h*)(p + 2 * off);
}

// B-fragment (32x16 f16): lane n = l&15 holds column n; 16 contiguous K values
// starting at (l>>4)*16.  p = row of B^T + k0.
__device__ __forceinline__ v16h load_b16(const _Float16* p, int hlf) {
  return *(const v16h*)(p + hlf * 16);
}

// ---------------------------------------------------------------------------
// Async global -> LDS copy (16 bytes per lane), CDNA5 ASYNCcnt path.
// ---------------------------------------------------------------------------
__device__ __forceinline__ void async_cp16(const _Float16* g, _Float16* l) {
#if ASYNC_LDS
  __builtin_amdgcn_global_load_async_to_lds_b128((gas_v4i*)g, (las_v4i*)l, 0, 0);
#else
  *(v8h*)l = *(const v8h*)g;
#endif
}

template <int N>
__device__ __forceinline__ void wait_async() {
#if ASYNC_LDS
#if __has_builtin(__builtin_amdgcn_s_wait_asynccnt)
  __builtin_amdgcn_s_wait_asynccnt(N);
#else
  asm volatile("s_wait_asynccnt %0" ::"n"(N) : "memory");
#endif
#endif
}

// ---------------------------------------------------------------------------
// K0: f32 -> f16 conversion kernels
// ---------------------------------------------------------------------------
__global__ __launch_bounds__(256) void cvt_pad_kernel(const float* __restrict__ src,
                                                      _Float16* __restrict__ dst,
                                                      int rows, int prows, int cols) {
  size_t i = (size_t)blockIdx.x * blockDim.x + threadIdx.x;
  size_t total = (size_t)prows * cols;
  if (i >= total) return;
  size_t r = i / cols;
  dst[i] = (r < (size_t)rows) ? (_Float16)src[i] : (_Float16)0.0f;
}

__global__ __launch_bounds__(256) void cvt_kernel(const float* __restrict__ src,
                                                  _Float16* __restrict__ dst, int n) {
  int i = blockIdx.x * blockDim.x + threadIdx.x;
  if (i < n) dst[i] = (_Float16)src[i];
}

// ---------------------------------------------------------------------------
// K1: QKV projections.  grid (LP/64, DIM/128, 3), block 256 (8 waves, 2x4).
//     z=0 -> qraw f32, z=1 -> kraw f32, z=2 -> vT f16 transposed.
// ---------------------------------------------------------------------------
__global__ __launch_bounds__(256) void gemm_qkv_kernel(
    const _Float16* __restrict__ xh,
    const _Float16* __restrict__ wq, const _Float16* __restrict__ wk,
    const _Float16* __restrict__ wv,
    const float* __restrict__ bq, const float* __restrict__ bk,
    const float* __restrict__ bv,
    float* __restrict__ qraw, float* __restrict__ kraw,
    _Float16* __restrict__ vT) {
  const int z = blockIdx.z;
  const _Float16* w   = (z == 0) ? wq : (z == 1) ? wk : wv;
  const float*    bia = (z == 0) ? bq : (z == 1) ? bk : bv;

  const int wave = threadIdx.x >> 5;
  const int lane = threadIdx.x & 31;
  const int n    = lane & 15;
  const int hlf  = lane >> 4;
  const int off  = hlf * 8;
  const int row0 = blockIdx.x * 64 + (wave >> 2) * 32;   // M
  const int col0 = blockIdx.y * 128 + (wave & 3) * 32;   // N

  v8f acc[2][2] = {};
  const _Float16* arow0 = xh + (size_t)(row0 + n) * DIM;
  const _Float16* arow1 = xh + (size_t)(row0 + 16 + n) * DIM;
  const _Float16* brow0 = w + (size_t)(col0 + n) * DIM;
  const _Float16* brow1 = w + (size_t)(col0 + 16 + n) * DIM;

  for (int k0 = 0; k0 < DIM; k0 += 32) {
    __builtin_prefetch(arow0 + k0 + 64, 0, 3);
    __builtin_prefetch(brow0 + k0 + 64, 0, 3);
    v16h a0 = load_a16(arow0 + k0, off);
    v16h a1 = load_a16(arow1 + k0, off);
    v16h b0 = load_b16(brow0 + k0, hlf);
    v16h b1 = load_b16(brow1 + k0, hlf);
    acc[0][0] = wmma_f16(a0, b0, acc[0][0]);
    acc[0][1] = wmma_f16(a0, b1, acc[0][1]);
    acc[1][0] = wmma_f16(a1, b0, acc[1][0]);
    acc[1][1] = wmma_f16(a1, b1, acc[1][1]);
  }

#pragma unroll
  for (int tm = 0; tm < 2; ++tm) {
#pragma unroll
    for (int tn = 0; tn < 2; ++tn) {
      const int colg = col0 + tn * 16 + n;
      const int rowg = row0 + tm * 16 + hlf * 8;
      const float bb = bia[colg];
      if (z < 2) {
        float* out = (z == 0) ? qraw : kraw;
#pragma unroll
        for (int r = 0; r < 8; ++r)
          out[(size_t)(rowg + r) * DIM + colg] = acc[tm][tn][r] + bb;
      } else {
        v8h h8;  // vT[feature][token]; 8 consecutive tokens -> 16B store
#pragma unroll
        for (int r = 0; r < 8; ++r) h8[r] = (_Float16)(acc[tm][tn][r] + bb);
        *(v8h*)(vT + (size_t)colg * LP + rowg) = h8;
      }
    }
  }
}

// ---------------------------------------------------------------------------
// K2: RMS-norm (over full DIM) + RoPE for q and k.  grid LP, block 256.
// ---------------------------------------------------------------------------
__global__ __launch_bounds__(256) void rms_rope_kernel(
    const float* __restrict__ qraw, const float* __restrict__ kraw,
    const float* __restrict__ gq, const float* __restrict__ gk,
    const float* __restrict__ freqs,
    _Float16* __restrict__ qh, _Float16* __restrict__ kh,
    const int* __restrict__ p_start) {
  const int t   = blockIdx.x;
  const int tid = threadIdx.x;
  const int lane = tid & 31, wave = tid >> 5;
  const float* qr = qraw + (size_t)t * DIM;
  const float* kr = kraw + (size_t)t * DIM;

  float sq = 0.f, sk = 0.f;
  for (int i = tid; i < DIM; i += 256) {
    float a = qr[i]; sq += a * a;
    float b = kr[i]; sk += b * b;
  }
#pragma unroll
  for (int o = 16; o >= 1; o >>= 1) {
    sq += __shfl_xor(sq, o, 32);
    sk += __shfl_xor(sk, o, 32);
  }
  __shared__ float redq[8], redk[8];
  if (lane == 0) { redq[wave] = sq; redk[wave] = sk; }
  __syncthreads();
  float tq = 0.f, tk = 0.f;
#pragma unroll
  for (int i = 0; i < 8; ++i) { tq += redq[i]; tk += redk[i]; }
  const float rq = rsqrtf(tq / (float)DIM + 1e-6f);
  const float rk = rsqrtf(tk / (float)DIM + 1e-6f);

  const int start = *p_start;
  const int fi = t / HW;
  const int hpos = (t % HW) / WN;
  const int wpos = t % WN;

  for (int p = tid; p < DIM / 2; p += 256) {
    const int head = p >> 6;      // 64 pairs per head
    const int ci   = p & 63;
    float ang;
    if (ci < C0)            ang = freqs[(size_t)(start + fi) * CC + ci];
    else if (ci < C0 + C1)  ang = freqs[(size_t)hpos * CC + ci];
    else                    ang = freqs[(size_t)wpos * CC + ci];
    const float cs = cosf(ang), sn = sinf(ang);
    const int idx = head * DH + 2 * ci;
    {
      const float re = qr[idx] * rq * gq[idx];
      const float im = qr[idx + 1] * rq * gq[idx + 1];
      qh[(size_t)t * DIM + idx]     = (_Float16)(re * cs - im * sn);
      qh[(size_t)t * DIM + idx + 1] = (_Float16)(re * sn + im * cs);
    }
    {
      const float re = kr[idx] * rk * gk[idx];
      const float im = kr[idx + 1] * rk * gk[idx + 1];
      kh[(size_t)t * DIM + idx]     = (_Float16)(re * cs - im * sn);
      kh[(size_t)t * DIM + idx + 1] = (_Float16)(re * sn + im * cs);
    }
  }
}

// ---------------------------------------------------------------------------
// K3: flash attention, frame-block causal mask.
//     grid (LP/64, NH), block 128 (4 waves; wave w owns q-tile blockbase+16w).
//     K/V 32-key tiles double-buffered in LDS via async global->LDS copies,
//     shared across all 4 waves.
// ---------------------------------------------------------------------------
__device__ __forceinline__ void stage_tiles(const _Float16* __restrict__ kh,
                                            const _Float16* __restrict__ vT,
                                            int head, int kbase,
                                            _Float16* kt, _Float16* vt, int tid) {
#pragma unroll
  for (int i = 0; i < 4; ++i) {
    const int c = tid + i * 128;            // 0..511 chunk id (16B chunks)
    const int row = c >> 4, col = c & 15;   // K tile: 32 tokens x 16 chunks
    async_cp16(kh + (size_t)(kbase + row) * DIM + head * DH + col * 8,
               kt + row * DH + col * 8);
    const int feat = c >> 1, part = c & 1;  // V tile: 256 feats x 2 chunks
    async_cp16(vT + (size_t)(head * DH + feat) * LP + kbase + part * 8,
               vt + feat * 32 + part * 8);
  }
}

__global__ __launch_bounds__(128) void attn_kernel(
    const _Float16* __restrict__ qh, const _Float16* __restrict__ kh,
    const _Float16* __restrict__ vT, _Float16* __restrict__ attnh,
    const int* __restrict__ p_local, const int* __restrict__ p_sink) {
  const int head = blockIdx.y;
  const int tid  = threadIdx.x;
  const int wave = tid >> 5, lane = tid & 31;
  const int n = lane & 15, hlf = lane >> 4, off = hlf * 8;
  const int blockbase = blockIdx.x * (16 * QW);
  const int qbase = blockbase + wave * 16;
  const int local = *p_local, sink = *p_sink;
  const float scale = 0.08838834764831845f;  // 1/sqrt(128)

  __shared__ __align__(32) _Float16 kt[2][32 * DH];   // [token][d] 8KB each
  __shared__ __align__(32) _Float16 vt[2][DH * 32];   // [d][token] 8KB each
  __shared__ float pls[QW][16 * 32];                  // per-wave P restage

  // resident Q A-fragments (16 x 128 = 4 K-chunks of 32)
  v16h qa[4];
  const _Float16* qrow = qh + (size_t)(qbase + n) * DIM + head * DH;
#pragma unroll
  for (int kk = 0; kk < 4; ++kk) qa[kk] = load_a16(qrow + kk * 32, off);

  v8f o[8] = {};
  float mrow[8], srow[8];
#pragma unroll
  for (int r = 0; r < 8; ++r) { mrow[r] = -1e30f; srow[r] = 0.f; }

  const int qmaxtok = (blockbase + 16 * QW - 1 < LTOK) ? (blockbase + 16 * QW - 1)
                                                       : (LTOK - 1);
  const int fimax = qmaxtok / HW;
  const int kend  = ((fimax + 1) * HW + 31) / 32;

  stage_tiles(kh, vT, head, 0, kt[0], vt[0], tid);

  for (int kb = 0; kb < kend; ++kb) {
    const int kbase = kb * 32;
    const int cur = kb & 1;
    if (kb + 1 < kend) {
      stage_tiles(kh, vT, head, kbase + 32, kt[cur ^ 1], vt[cur ^ 1], tid);
      wait_async<8>();   // previous batch (current buffer) complete
    } else {
      wait_async<0>();
    }
    __syncthreads();

    const _Float16* ktc = kt[cur];
    const _Float16* vtc = vt[cur];

    // ---- S = Q K^T for 32 keys (two 16x16 accumulators) ----
    v8f s0 = {}, s1 = {};
#pragma unroll
    for (int kk = 0; kk < 4; ++kk) {
      v16h b0 = *(const v16h*)(ktc + n * DH + kk * 32 + hlf * 16);
      v16h b1 = *(const v16h*)(ktc + (16 + n) * DH + kk * 32 + hlf * 16);
      s0 = wmma_f16(qa[kk], b0, s0);
      s1 = wmma_f16(qa[kk], b1, s1);
    }
    // ---- mask + online softmax (rows owned by this lane's half) ----
    const int j0 = kbase + n, j1 = j0 + 16;
    const int fj0 = j0 / HW, fj1 = j1 / HW;
#pragma unroll
    for (int r = 0; r < 8; ++r) {
      const int qtok = qbase + hlf * 8 + r;
      const int fi = qtok / HW;
      const bool a0 = (fj0 <= fi) && ((fi - fj0 < local) || (fj0 < sink));
      const bool a1 = (fj1 <= fi) && ((fi - fj1 < local) || (fj1 < sink));
      float x0 = a0 ? s0[r] * scale : -1e30f;
      float x1 = a1 ? s1[r] * scale : -1e30f;
      float rm = fmaxf(x0, x1);
#pragma unroll
      for (int oo = 8; oo >= 1; oo >>= 1) rm = fmaxf(rm, __shfl_xor(rm, oo, 16));
      const float mn = fmaxf(mrow[r], rm);
      const float corr = __expf(mrow[r] - mn);
      mrow[r] = mn;
      const float e0 = __expf(x0 - mn);
      const float e1 = __expf(x1 - mn);
      s0[r] = e0; s1[r] = e1;
      float rs = e0 + e1;
#pragma unroll
      for (int oo = 8; oo >= 1; oo >>= 1) rs += __shfl_xor(rs, oo, 16);
      srow[r] = srow[r] * corr + rs;
#pragma unroll
      for (int t2 = 0; t2 < 8; ++t2) o[t2][r] *= corr;
    }
    // ---- restage P (C-layout) into A-fragment layout via per-wave LDS ----
    float* pw = pls[wave];
#pragma unroll
    for (int r = 0; r < 8; ++r) {
      pw[(hlf * 8 + r) * 32 + n]      = s0[r];
      pw[(hlf * 8 + r) * 32 + 16 + n] = s1[r];
    }
    v16h pa;
#pragma unroll
    for (int i = 0; i < 8; ++i) {
      pa[i]     = (_Float16)pw[n * 32 + 2 * off + i];
      pa[i + 8] = (_Float16)pw[n * 32 + 2 * off + 8 + i];
    }
    // ---- O += P @ V (vt rows are ready-made B-fragments) ----
#pragma unroll
    for (int t2 = 0; t2 < 8; ++t2) {
      v16h vb = *(const v16h*)(vtc + (t2 * 16 + n) * 32 + hlf * 16);
      o[t2] = wmma_f16(pa, vb, o[t2]);
    }
    __syncthreads();  // everyone done with this buffer before it is refilled
  }

  // ---- normalize + store f16 ----
#pragma unroll
  for (int t2 = 0; t2 < 8; ++t2) {
    const int colg = head * DH + t2 * 16 + n;
#pragma unroll
    for (int r = 0; r < 8; ++r) {
      const float val = o[t2][r] / srow[r];
      attnh[(size_t)(qbase + hlf * 8 + r) * DIM + colg] = (_Float16)val;
    }
  }
}

// ---------------------------------------------------------------------------
// K4: output projection  out = attn @ Wo^T + bo  (f32, guarded to LTOK rows)
// ---------------------------------------------------------------------------
__global__ __launch_bounds__(256) void gemm_out_kernel(
    const _Float16* __restrict__ ah, const _Float16* __restrict__ woh,
    const float* __restrict__ bo, float* __restrict__ out) {
  const int wave = threadIdx.x >> 5;
  const int lane = threadIdx.x & 31;
  const int n = lane & 15, hlf = lane >> 4, off = hlf * 8;
  const int row0 = blockIdx.x * 64 + (wave >> 2) * 32;
  const int col0 = blockIdx.y * 128 + (wave & 3) * 32;

  v8f acc[2][2] = {};
  const _Float16* arow0 = ah + (size_t)(row0 + n) * DIM;
  const _Float16* arow1 = ah + (size_t)(row0 + 16 + n) * DIM;
  const _Float16* brow0 = woh + (size_t)(col0 + n) * DIM;
  const _Float16* brow1 = woh + (size_t)(col0 + 16 + n) * DIM;

  for (int k0 = 0; k0 < DIM; k0 += 32) {
    __builtin_prefetch(arow0 + k0 + 64, 0, 3);
    __builtin_prefetch(brow0 + k0 + 64, 0, 3);
    v16h a0 = load_a16(arow0 + k0, off);
    v16h a1 = load_a16(arow1 + k0, off);
    v16h b0 = load_b16(brow0 + k0, hlf);
    v16h b1 = load_b16(brow1 + k0, hlf);
    acc[0][0] = wmma_f16(a0, b0, acc[0][0]);
    acc[0][1] = wmma_f16(a0, b1, acc[0][1]);
    acc[1][0] = wmma_f16(a1, b0, acc[1][0]);
    acc[1][1] = wmma_f16(a1, b1, acc[1][1]);
  }

#pragma unroll
  for (int tm = 0; tm < 2; ++tm) {
#pragma unroll
    for (int tn = 0; tn < 2; ++tn) {
      const int colg = col0 + tn * 16 + n;
      const int rowg = row0 + tm * 16 + hlf * 8;
      const float bb = bo[colg];
#pragma unroll
      for (int r = 0; r < 8; ++r) {
        const int rr = rowg + r;
        if (rr < LTOK) out[(size_t)rr * DIM + colg] = acc[tm][tn][r] + bb;
      }
    }
  }
}

// ---------------------------------------------------------------------------
extern "C" void kernel_launch(void* const* d_in, const int* in_sizes, int n_in,
                              void* d_out, int out_size, void* d_ws, size_t ws_size,
                              hipStream_t stream) {
  (void)in_sizes; (void)n_in; (void)out_size; (void)ws_size;

  const float* x     = (const float*)d_in[0];
  const float* freqs = (const float*)d_in[1];
  const float* Wq = (const float*)d_in[2];  const float* bq = (const float*)d_in[3];
  const float* Wk = (const float*)d_in[4];  const float* bk = (const float*)d_in[5];
  const float* Wv = (const float*)d_in[6];  const float* bv = (const float*)d_in[7];
  const float* Wo = (const float*)d_in[8];  const float* bo = (const float*)d_in[9];
  const float* gq = (const float*)d_in[10]; const float* gk = (const float*)d_in[11];
  const int* p_local = (const int*)d_in[16];
  const int* p_sink  = (const int*)d_in[17];
  const int* p_start = (const int*)d_in[18];
  float* out = (float*)d_out;

  char* ws = (char*)d_ws;
  _Float16* xh    = (_Float16*)(ws + OFF_XH);
  _Float16* wqh   = (_Float16*)(ws + OFF_WQH);
  _Float16* wkh   = (_Float16*)(ws + OFF_WKH);
  _Float16* wvh   = (_Float16*)(ws + OFF_WVH);
  _Float16* woh   = (_Float16*)(ws + OFF_WOH);
  float*    qraw  = (float*)(ws + OFF_QRAW);
  float*    kraw  = (float*)(ws + OFF_KRAW);
  _Float16* qh    = (_Float16*)(ws + OFF_QH);
  _Float16* kh    = (_Float16*)(ws + OFF_KH);
  _Float16* vT    = (_Float16*)(ws + OFF_VT);
  _Float16* attnh = (_Float16*)(ws + OFF_ATTNH);

  // K0: conversions
  {
    size_t tot = (size_t)LP * DIM;
    cvt_pad_kernel<<<(unsigned)((tot + 255) / 256), 256, 0, stream>>>(x, xh, LTOK, LP, DIM);
    int wn = DIM * DIM;
    unsigned wb = (unsigned)((wn + 255) / 256);
    cvt_kernel<<<wb, 256, 0, stream>>>(Wq, wqh, wn);
    cvt_kernel<<<wb, 256, 0, stream>>>(Wk, wkh, wn);
    cvt_kernel<<<wb, 256, 0, stream>>>(Wv, wvh, wn);
    cvt_kernel<<<wb, 256, 0, stream>>>(Wo, woh, wn);
  }
  // K1: QKV projections (z selects q/k/v)
  gemm_qkv_kernel<<<dim3(LP / 64, DIM / 128, 3), 256, 0, stream>>>(
      xh, wqh, wkh, wvh, bq, bk, bv, qraw, kraw, vT);
  // K2: RMS-norm + RoPE
  rms_rope_kernel<<<LP, 256, 0, stream>>>(qraw, kraw, gq, gk, freqs, qh, kh, p_start);
  // K3: attention (4 q-tiles per block share async-staged K/V tiles)
  attn_kernel<<<dim3(LP / (16 * QW), NH), 128, 0, stream>>>(qh, kh, vT, attnh,
                                                            p_local, p_sink);
  // K4: output projection
  gemm_out_kernel<<<dim3(LP / 64, DIM / 128), 256, 0, stream>>>(attnh, woh, bo, out);
}
